// GroupQueryAttention_10187662426352
// MI455X (gfx1250) — matrix operand order repaired and checked by reading, not compiled, not run on hardware
//
#include <hip/hip_runtime.h>

typedef _Float16 half_t;
typedef __attribute__((ext_vector_type(16))) _Float16 v16h;
typedef __attribute__((ext_vector_type(8)))  _Float16 h8;
typedef __attribute__((ext_vector_type(8)))  float    v8f;

#define BB  2
#define SS  2048
#define EE  2048
#define HH  32
#define GG  8
#define HDD 64
#define GSZ 4
#define KVD 512
#define MMR (BB*SS)

// ---- fragment layouts (v_wmma_f32_16x16x32_f16) -----------------------------
// A (16x32 f16): lane L holds row M=L&15; halves: K = hi*16 + {0..15} (contiguous, same order as B)
// B (32x16 f16): lane L holds col N=L&15; halves: K = hi*16 + {0..15}
// C (16x16 f32): lane L, vgpr r -> (M = r + 8*hi, N = L&15)
__device__ inline v16h ldfrag(const half_t* p0, const half_t* p1) {
  union { v16h v; h8 h[2]; } u;
  u.h[0] = *(const h8*)p0;
  u.h[1] = *(const h8*)p1;
  return u.v;
}

// LDS-pinned accessors: amdgcn generic->local addrspacecast == take low 32 bits,
// so routing through u32 keeps these on the DS path (ds_load_b128/ds_store_b16).
__device__ inline unsigned lds_off(const void* p) {
  return (unsigned)(unsigned long long)p;
}
__device__ inline v16h ldfrag_lds(unsigned a) {
  const __attribute__((address_space(3))) h8* p0 =
      (const __attribute__((address_space(3))) h8*)a;
  const __attribute__((address_space(3))) h8* p1 =
      (const __attribute__((address_space(3))) h8*)(a + 16);
  union { v16h v; h8 h[2]; } u;
  u.h[0] = *p0;
  u.h[1] = *p1;
  return u.v;
}
__device__ inline void st_lds_h16(unsigned a, half_t v) {
  *(__attribute__((address_space(3))) half_t*)a = v;
}

__device__ inline float redmax16(float v) {
  v = fmaxf(v, __shfl_xor(v, 1, 16));
  v = fmaxf(v, __shfl_xor(v, 2, 16));
  v = fmaxf(v, __shfl_xor(v, 4, 16));
  v = fmaxf(v, __shfl_xor(v, 8, 16));
  return v;
}
__device__ inline float redsum16(float v) {
  v += __shfl_xor(v, 1, 16);
  v += __shfl_xor(v, 2, 16);
  v += __shfl_xor(v, 4, 16);
  v += __shfl_xor(v, 8, 16);
  return v;
}

// ---- elementwise helpers ----------------------------------------------------
__global__ void k_f32tof16(const float* __restrict__ in, half_t* __restrict__ out, int n) {
  int i = blockIdx.x * blockDim.x + threadIdx.x;
  if (i < n) out[i] = (half_t)in[i];
}

// W[K][N] (f32) -> Wt[N][K] (f16)
__global__ void k_transpose(const float* __restrict__ W, half_t* __restrict__ Wt, int K, int N) {
  int i = blockIdx.x * blockDim.x + threadIdx.x;
  if (i < K * N) {
    int k = i / N, n = i - k * N;
    Wt[(long)n * K + k] = (half_t)W[i];
  }
}

// v[b*S+t][g*64+hd] -> vT[b][g][hd][t]
__global__ void k_build_vT(const half_t* __restrict__ v, half_t* __restrict__ vT) {
  int i = blockIdx.x * blockDim.x + threadIdx.x;
  if (i < MMR * KVD) {
    int t = i >> 9; int rem = i & 511;
    int g = rem >> 6; int hd = rem & 63;
    int b = t / SS;  int s = t - b * SS;
    vT[(((long)(b * GG + g) * HDD + hd) * SS) + s] = v[i];
  }
}

// ---- WMMA GEMM: C[M,N] = A[M,K] * Wt^T + bias; one wave computes 16x128 -----
template <bool OUT16>
__global__ __launch_bounds__(256, 1)
void k_gemm(const half_t* __restrict__ A, const half_t* __restrict__ Wt,
            const float* __restrict__ bias, void* __restrict__ outp,
            int M, int N, int K) {
  const int lane = threadIdx.x & 31;
  const int nl = lane & 15;
  const int hi = lane >> 4;
  const int gw = (blockIdx.x * blockDim.x + threadIdx.x) >> 5;
  const int tilesN = N >> 7;
  const int m0 = (gw / tilesN) << 4;
  const int n0 = (gw % tilesN) << 7;
  if (m0 >= M) return;

  v8f acc[8] = {};
  const half_t* ap = A + (long)(m0 + nl) * K + hi * 16;
  const half_t* bp[8];
#pragma unroll
  for (int c = 0; c < 8; c++)
    bp[c] = Wt + (long)(n0 + c * 16 + nl) * K + hi * 16;

  for (int k0 = 0; k0 < K; k0 += 32) {
    v16h a = ldfrag(ap, ap + 8);
    __builtin_prefetch(ap + 64, 0, 3);
    __builtin_prefetch(bp[0] + 64, 0, 3);
#pragma unroll
    for (int c = 0; c < 8; c++) {
      v16h bf = ldfrag(bp[c], bp[c] + 8);
      acc[c] = __builtin_amdgcn_wmma_f32_16x16x32_f16(false, a, false, bf,
                                                      (short)0, acc[c], false, false);
      bp[c] += 32;
    }
    ap += 32;
  }
#pragma unroll
  for (int c = 0; c < 8; c++) {
    int n = n0 + c * 16 + nl;
    float bv = bias[n];
#pragma unroll
    for (int r = 0; r < 8; r++) {
      int row = m0 + r + 8 * hi;
      float val = acc[c][r] + bv;
      if (OUT16) ((half_t*)outp)[(long)row * N + n] = (half_t)val;
      else       ((float*)outp)[(long)row * N + n] = val;
    }
  }
}

// ---- causal flash attention: one wave = 16 query rows of one head -----------
__global__ __launch_bounds__(256, 1)
void k_flash(const half_t* __restrict__ q, const half_t* __restrict__ k,
             const half_t* __restrict__ vT, half_t* __restrict__ o) {
  __shared__ half_t ldsP[8][16 * 32];   // per-wave P tile [query 16][key 32]
  const int lane = threadIdx.x & 31;
  const int nl = lane & 15, hi = lane >> 4;
  const int wid = threadIdx.x >> 5;
  const int gw = blockIdx.x * 8 + wid;
  const int qtiles = SS / 16;
  if (gw >= BB * HH * qtiles) return;
  int b = gw / (HH * qtiles);
  int rem = gw - b * (HH * qtiles);
  int h = rem / qtiles;
  int qt = rem - h * qtiles;
  int g = h / GSZ;
  int qbase = qt * 16;

  // preload Q fragments (HD=64 -> two 32-wide K slices)
  const half_t* qrow = q + (long)(b * SS + qbase + nl) * EE + h * HDD;
  v16h qa0 = ldfrag(qrow + hi * 16,      qrow + hi * 16 + 8);
  v16h qa1 = ldfrag(qrow + 32 + hi * 16, qrow + 32 + hi * 16 + 8);

  v8f O[4] = {};
  float mrow[8], lrow[8];
#pragma unroll
  for (int r = 0; r < 8; r++) { mrow[r] = -1e30f; lrow[r] = 0.f; }

  const float scl = 0.125f;  // 1/sqrt(64)
  const int kend = qbase + 16;

  // hoisted, incremented pointers (no per-iteration 64-bit muls)
  const half_t* kptr0 = k + (long)(b * SS + nl) * KVD + g * HDD + hi * 16;
  const half_t* kptr1 = kptr0 + (long)16 * KVD;
  const half_t* vptr[4];
#pragma unroll
  for (int c = 0; c < 4; c++)
    vptr[c] = vT + (((long)(b * GG + g) * HDD + c * 16 + nl) * SS) + hi * 16;
  const unsigned pwr = lds_off(&ldsP[wid][(8 * hi) * 32 + nl]);   // write base, bytes
  const unsigned prd = lds_off(&ldsP[wid][nl * 32 + hi * 16]);    // A-frag read base
  int d0 = nl - qbase - 8 * hi;                                   // causal: mask when d0 > r

  for (int kb = 0; kb < kend; kb += 32) {
    v8f c0 = {}, c1 = {};
    __builtin_prefetch(kptr0 + (long)32 * KVD, 0, 3);
    {
      v16h b0 = ldfrag(kptr0,      kptr0 + 8);
      c0 = __builtin_amdgcn_wmma_f32_16x16x32_f16(false, qa0, false, b0, (short)0, c0, false, false);
      v16h b1 = ldfrag(kptr0 + 32, kptr0 + 40);
      c0 = __builtin_amdgcn_wmma_f32_16x16x32_f16(false, qa1, false, b1, (short)0, c0, false, false);
      v16h b2 = ldfrag(kptr1,      kptr1 + 8);
      c1 = __builtin_amdgcn_wmma_f32_16x16x32_f16(false, qa0, false, b2, (short)0, c1, false, false);
      v16h b3 = ldfrag(kptr1 + 32, kptr1 + 40);
      c1 = __builtin_amdgcn_wmma_f32_16x16x32_f16(false, qa1, false, b3, (short)0, c1, false, false);
    }
    kptr0 += (long)32 * KVD;
    kptr1 += (long)32 * KVD;

    // scale + causal mask in place (compare against immediate r)
    const int d1 = d0 + 16;
#pragma unroll
    for (int r = 0; r < 8; r++) {
      float s0 = c0[r] * scl; if (d0 > r) s0 = -1e30f; c0[r] = s0;
      float s1 = c1[r] * scl; if (d1 > r) s1 = -1e30f; c1[r] = s1;
    }
    d0 += 32;

    // online softmax per row (row stats uniform across each 16-lane half)
#pragma unroll
    for (int r = 0; r < 8; r++) {
      float bm = redmax16(fmaxf(c0[r], c1[r]));
      float mnew = fmaxf(mrow[r], bm);
      float corr = __expf(mrow[r] - mnew);
      float p0 = __expf(c0[r] - mnew);
      float p1 = __expf(c1[r] - mnew);
      lrow[r] = lrow[r] * corr + redsum16(p0 + p1);
      mrow[r] = mnew;
#pragma unroll
      for (int c = 0; c < 4; c++) O[c][r] *= corr;
      st_lds_h16(pwr + (unsigned)(r * 32) * 2u,        (half_t)p0);
      st_lds_h16(pwr + (unsigned)(r * 32 + 16) * 2u,   (half_t)p1);
    }

    // re-layout P (C layout -> A layout) through LDS, then O += P @ V
    v16h pa = ldfrag_lds(prd);
#pragma unroll
    for (int c = 0; c < 4; c++) {
      v16h bf = ldfrag(vptr[c], vptr[c] + 8);
      O[c] = __builtin_amdgcn_wmma_f32_16x16x32_f16(false, pa, false, bf,
                                                    (short)0, O[c], false, false);
      vptr[c] += 32;
    }
  }

  // normalize and write attention output in [b, s, h*HD+hd] layout (f16)
#pragma unroll
  for (int c = 0; c < 4; c++) {
    int col = h * HDD + c * 16 + nl;
#pragma unroll
    for (int r = 0; r < 8; r++) {
      int row = qbase + r + 8 * hi;
      float val = O[c][r] / lrow[r];
      o[(long)(b * SS + row) * EE + col] = (half_t)val;
    }
  }
}

// ---- launch -----------------------------------------------------------------
extern "C" void kernel_launch(void* const* d_in, const int* in_sizes, int n_in,
                              void* d_out, int out_size, void* d_ws, size_t ws_size,
                              hipStream_t stream) {
  (void)in_sizes; (void)n_in; (void)out_size;
  const float* x  = (const float*)d_in[0];
  const float* Wq = (const float*)d_in[1];
  const float* bq = (const float*)d_in[2];
  const float* Wk = (const float*)d_in[3];
  const float* bk = (const float*)d_in[4];
  const float* Wv = (const float*)d_in[5];
  const float* bv = (const float*)d_in[6];
  const float* Wo = (const float*)d_in[7];
  const float* bo = (const float*)d_in[8];
  float* out = (float*)d_out;

  char* w = (char*)d_ws;
  size_t off = 0;
  auto alloc = [&](size_t bytes) -> char* {
    char* p = w + off;
    off += (bytes + 255) & ~(size_t)255;
    return p;
  };
  half_t* x_h  = (half_t*)alloc((size_t)MMR * EE * 2);
  half_t* Wqt  = (half_t*)alloc((size_t)EE * EE * 2);
  half_t* Wkt  = (half_t*)alloc((size_t)EE * KVD * 2);
  half_t* Wvt  = (half_t*)alloc((size_t)EE * KVD * 2);
  half_t* Wot  = (half_t*)alloc((size_t)EE * EE * 2);
  half_t* q_h  = (half_t*)alloc((size_t)MMR * EE * 2);
  half_t* k_h  = (half_t*)alloc((size_t)MMR * KVD * 2);
  half_t* v_h  = (half_t*)alloc((size_t)MMR * KVD * 2);
  half_t* vT_h = (half_t*)alloc((size_t)MMR * KVD * 2);
  half_t* a_h  = (half_t*)alloc((size_t)MMR * EE * 2);
  if (off > ws_size) return;

  const int T = 256;
  k_f32tof16<<<(MMR * EE + T - 1) / T, T, 0, stream>>>(x, x_h, MMR * EE);
  k_transpose<<<(EE * EE + T - 1) / T, T, 0, stream>>>(Wq, Wqt, EE, EE);
  k_transpose<<<(EE * KVD + T - 1) / T, T, 0, stream>>>(Wk, Wkt, EE, KVD);
  k_transpose<<<(EE * KVD + T - 1) / T, T, 0, stream>>>(Wv, Wvt, EE, KVD);
  k_transpose<<<(EE * EE + T - 1) / T, T, 0, stream>>>(Wo, Wot, EE, EE);

  {
    int waves = (MMR / 16) * (EE / 128);  // 4096 waves -> 512 blocks
    k_gemm<true><<<waves * 32 / T, T, 0, stream>>>(x_h, Wqt, bq, q_h, MMR, EE, EE);
  }
  {
    int waves = (MMR / 16) * (KVD / 128); // 1024 waves -> 128 blocks
    k_gemm<true><<<waves * 32 / T, T, 0, stream>>>(x_h, Wkt, bk, k_h, MMR, KVD, EE);
    k_gemm<true><<<waves * 32 / T, T, 0, stream>>>(x_h, Wvt, bv, v_h, MMR, KVD, EE);
  }
  k_build_vT<<<(MMR * KVD + T - 1) / T, T, 0, stream>>>(v_h, vT_h);
  {
    int waves = BB * HH * (SS / 16);      // 8192 waves -> 1024 blocks
    k_flash<<<waves / 8, T, 0, stream>>>(q_h, k_h, vT_h, a_h);
  }
  {
    int waves = (MMR / 16) * (EE / 128);
    k_gemm<false><<<waves * 32 / T, T, 0, stream>>>(a_h, Wot, bo, out, MMR, EE, EE);
  }
}
